// GameStateMoERouter_55705725829209
// MI455X (gfx1250) — matrix operand order repaired and checked by reading, not compiled, not run on hardware
//
#include <hip/hip_runtime.h>
#include <cstdint>
#include <cstddef>

#define B_  16384
#define V_  1024
#define MM_ 128
#define H_  1024
#define E_  8
#define D_  256

typedef __attribute__((ext_vector_type(16))) __bf16 v16bf;
typedef __attribute__((ext_vector_type(8)))  float  v8f;
typedef __attribute__((ext_vector_type(4)))  int    vi4;

union FragU { uint4 q[2]; v16bf v; };

__device__ __forceinline__ unsigned short f2bf(float f) {
  unsigned u = __float_as_uint(f);
  u += 0x7FFFu + ((u >> 16) & 1u);
  return (unsigned short)(u >> 16);
}
__device__ __forceinline__ float bf2f(unsigned short h) {
  return __uint_as_float(((unsigned)h) << 16);
}

// CDNA5 async global->LDS path (ASYNCcnt) when the toolchain declares it.
#if __has_builtin(__builtin_amdgcn_global_load_async_to_lds_b128)
#define HAVE_ASYNC_LDS 1
#else
#define HAVE_ASYNC_LDS 0
#endif

#if __has_builtin(__builtin_amdgcn_s_wait_asynccnt)
#define WAIT_ASYNC0() __builtin_amdgcn_s_wait_asynccnt(0)
#else
#define WAIT_ASYNC0() asm volatile("s_wait_asynccnt 0" ::: "memory")
#endif

// ---------------------------------------------------------------------------
// Tiled WMMA GEMM: C[M,N](f32) = A[M,K](bf16 row-major) * Bt[N,K](bf16)^T + bias
// Tile 256x128x32, 256 threads = 8 waves (4x2), wave tile 64x64 (4x4 WMMAs).
// Double-buffered LDS (one barrier per K-tile); async global->LDS if available.
// ---------------------------------------------------------------------------
#define BM 256
#define BN 128
#define BK 32
#define LDSK 40   // padded LDS row stride (elements); 80B keeps 16B alignment

__global__ __launch_bounds__(256)
void gemm_bf16_wmma(const unsigned short* __restrict__ A, int lda,
                    const unsigned short* __restrict__ Bt, int ldb,
                    const float* __restrict__ bias,
                    float* __restrict__ C, int ldc, int Kdim) {
  __shared__ unsigned short As[2][BM * LDSK];
  __shared__ unsigned short Bs[2][BN * LDSK];

  const int tid  = threadIdx.x;
  const int wave = tid >> 5;
  const int lane = tid & 31;
  const int half = lane >> 4;
  const int lr   = lane & 15;
  const int wr   = wave >> 1;    // 0..3 (64-row block)
  const int wc   = wave & 1;     // 0..1 (64-col block)
  const int m0   = blockIdx.y * BM;
  const int n0   = blockIdx.x * BN;

  v8f zero = {0.f,0.f,0.f,0.f,0.f,0.f,0.f,0.f};
  v8f acc[4][4];
#pragma unroll
  for (int i = 0; i < 4; ++i)
#pragma unroll
    for (int j = 0; j < 4; ++j) acc[i][j] = zero;

  // Tile staging assignment: A 256x32 -> 1 thread/row (4x b128);
  //                          B 128x32 -> 2 threads/row (2x b128 each).
  const int aRow = tid;
  const int bRow = tid >> 1;
  const int bCol = (tid & 1) * 16;
  const unsigned short* gA = A  + (size_t)(m0 + aRow) * lda;
  const unsigned short* gB = Bt + (size_t)(n0 + bRow) * ldb + bCol;

#if HAVE_ASYNC_LDS
  auto issue_tile = [&](int kt, int buf) {
    const unsigned short* a = gA + kt;
    unsigned short* lA = &As[buf][aRow * LDSK];
#pragma unroll
    for (int q = 0; q < 4; ++q)
      __builtin_amdgcn_global_load_async_to_lds_b128((vi4*)(a + q * 8), (vi4*)(lA + q * 8), 0, 0);
    const unsigned short* b = gB + kt;
    unsigned short* lB = &Bs[buf][bRow * LDSK + bCol];
    __builtin_amdgcn_global_load_async_to_lds_b128((vi4*)(b),     (vi4*)(lB),     0, 0);
    __builtin_amdgcn_global_load_async_to_lds_b128((vi4*)(b + 8), (vi4*)(lB + 8), 0, 0);
  };
#else
  uint4 ra[4], rb[2];
  auto gload = [&](int kt) {
#pragma unroll
    for (int q = 0; q < 4; ++q) ra[q] = *(const uint4*)(gA + kt + q * 8);
    rb[0] = *(const uint4*)(gB + kt);
    rb[1] = *(const uint4*)(gB + kt + 8);
  };
  auto lstore = [&](int buf) {
#pragma unroll
    for (int q = 0; q < 4; ++q) *(uint4*)&As[buf][aRow * LDSK + q * 8] = ra[q];
    *(uint4*)&Bs[buf][bRow * LDSK + bCol]     = rb[0];
    *(uint4*)&Bs[buf][bRow * LDSK + bCol + 8] = rb[1];
  };
#endif

  auto compute = [&](int buf) {
    FragU af[4], bfr[4];
#pragma unroll
    for (int i = 0; i < 4; ++i) {
      // 16-bit A layout: lane<16 -> K 0..7 & 16..23 ; lane>=16 -> K 8..15 & 24..31
      const unsigned short* ap = &As[buf][(wr * 64 + i * 16 + lr) * LDSK + half * 8];
      af[i].q[0] = *(const uint4*)(ap);
      af[i].q[1] = *(const uint4*)(ap + 16);
    }
#pragma unroll
    for (int j = 0; j < 4; ++j) {
      // B 32x16: lane<16 -> K 0..7 & 16..23 of column lr ; lane>=16 -> K 8..15 & 24..31
      const unsigned short* bp = &Bs[buf][(wc * 64 + j * 16 + lr) * LDSK + half * 8];
      bfr[j].q[0] = *(const uint4*)(bp);
      bfr[j].q[1] = *(const uint4*)(bp + 16);
    }
#pragma unroll
    for (int i = 0; i < 4; ++i)
#pragma unroll
      for (int j = 0; j < 4; ++j)
        acc[i][j] = __builtin_amdgcn_wmma_f32_16x16x32_bf16(
            false, af[i].v, false, bfr[j].v, (short)0, acc[i][j], false, false);
  };

  // ---- software pipeline: prologue fills buffer 0 ----
#if HAVE_ASYNC_LDS
  issue_tile(0, 0);
  WAIT_ASYNC0();
  __syncthreads();
  int cur = 0;
  for (int kt = 0;;) {
    const int nxt = kt + BK;
    const bool have_next = nxt < Kdim;
    if (have_next) issue_tile(nxt, cur ^ 1);   // fetch next while computing
    compute(cur);
    if (!have_next) break;
    WAIT_ASYNC0();
    __syncthreads();                           // one barrier per K-tile
    cur ^= 1;
    kt = nxt;
  }
#else
  gload(0);
  lstore(0);
  __syncthreads();
  int cur = 0;
  for (int kt = 0;;) {
    const int nxt = kt + BK;
    const bool have_next = nxt < Kdim;
    if (have_next) {
      gload(nxt);                              // overlap with compute below
      __builtin_prefetch(gA + nxt + BK, 0, 3);
      __builtin_prefetch(gB + nxt + BK, 0, 3);
    }
    compute(cur);
    if (!have_next) break;
    lstore(cur ^ 1);                           // other buffer: no wait needed
    __syncthreads();                           // one barrier per K-tile
    cur ^= 1;
    kt = nxt;
  }
#endif

  // Epilogue: C/D layout -> lane half selects M base (0 or 8), vgpr v adds to M.
#pragma unroll
  for (int i = 0; i < 4; ++i)
#pragma unroll
    for (int j = 0; j < 4; ++j) {
      const int gcol = n0 + wc * 64 + j * 16 + lr;
      const float bv = bias ? bias[gcol] : 0.f;
#pragma unroll
      for (int v = 0; v < 8; ++v) {
        const int grow = m0 + wr * 64 + i * 16 + half * 8 + v;
        C[(size_t)grow * ldc + gcol] = acc[i][j][v] + bv;
      }
    }
}

// ---------------------------------------------------------------------------
// Row-wise LayerNorm + exact GELU; f32 in, bf16 out (strided output).
// ---------------------------------------------------------------------------
__global__ __launch_bounds__(256)
void ln_gelu_kernel(const float* __restrict__ X, int N,
                    const float* __restrict__ g, const float* __restrict__ b,
                    unsigned short* __restrict__ Y, int ldy) {
  const int row = blockIdx.x;
  const int tid = threadIdx.x;
  const float* x = X + (size_t)row * N;
  float s = 0.f, s2 = 0.f;
  for (int i = tid; i < N; i += 256) { float v = x[i]; s += v; s2 += v * v; }
  __shared__ float rs[256], rq[256];
  rs[tid] = s; rq[tid] = s2;
  __syncthreads();
  for (int off = 128; off > 0; off >>= 1) {
    if (tid < off) { rs[tid] += rs[tid + off]; rq[tid] += rq[tid + off]; }
    __syncthreads();
  }
  const float mean = rs[0] / (float)N;
  const float var  = rq[0] / (float)N - mean * mean;
  const float inv  = rsqrtf(var + 1e-5f);
  for (int i = tid; i < N; i += 256) {
    float v = (x[i] - mean) * inv * g[i] + b[i];
    v = 0.5f * v * (1.f + erff(v * 0.70710678118654752f));
    Y[(size_t)row * ldy + i] = f2bf(v);
  }
}

// ---------------------------------------------------------------------------
// f32 -> bf16 conversion (grid-stride) and transposing variant (W -> W^T).
// ---------------------------------------------------------------------------
__global__ void cvt_kernel(const float* __restrict__ s, unsigned short* __restrict__ d, long n) {
  long i = (long)blockIdx.x * blockDim.x + threadIdx.x;
  long stride = (long)gridDim.x * blockDim.x;
  for (; i < n; i += stride) d[i] = f2bf(s[i]);
}
__global__ void cvtT_kernel(const float* __restrict__ s, unsigned short* __restrict__ d,
                            int K, int N) {
  long total = (long)K * N;
  long i = (long)blockIdx.x * blockDim.x + threadIdx.x;
  long stride = (long)gridDim.x * blockDim.x;
  for (; i < total; i += stride) {
    int k = (int)(i / N), n = (int)(i % N);
    d[(size_t)n * K + k] = f2bf(s[i]);
  }
}

// ---------------------------------------------------------------------------
// Small-N GEMM (N<=8): one thread per row, bf16 A, f32 W[K,N].
// ---------------------------------------------------------------------------
template <int N>
__global__ __launch_bounds__(256)
void small_gemm(const unsigned short* __restrict__ A, int Kdim,
                const float* __restrict__ W, const float* __restrict__ bias,
                float* __restrict__ out) {
  const int row = blockIdx.x * blockDim.x + threadIdx.x;
  if (row >= B_) return;
  float acc[N];
#pragma unroll
  for (int n = 0; n < N; ++n) acc[n] = bias[n];
  const unsigned short* a = A + (size_t)row * Kdim;
  for (int k = 0; k < Kdim; ++k) {
    float av = bf2f(a[k]);
#pragma unroll
    for (int n = 0; n < N; ++n) acc[n] += av * W[k * N + n];
  }
#pragma unroll
  for (int n = 0; n < N; ++n) out[(size_t)row * N + n] = acc[n];
}

// ---------------------------------------------------------------------------
// Gate: softmax over E=8, top-2, renormalize, scatter to dense weights.
// ---------------------------------------------------------------------------
__global__ __launch_bounds__(256)
void gate_kernel(const float* __restrict__ logits, float* __restrict__ out_gw,
                 float* __restrict__ out_tki, float* __restrict__ out_tkw,
                 float* __restrict__ wfull) {
  const int row = blockIdx.x * blockDim.x + threadIdx.x;
  if (row >= B_) return;
  float z[E_], mx = -1e30f;
#pragma unroll
  for (int e = 0; e < E_; ++e) { z[e] = logits[(size_t)row * E_ + e]; mx = fmaxf(mx, z[e]); }
  float sum = 0.f;
#pragma unroll
  for (int e = 0; e < E_; ++e) { z[e] = __expf(z[e] - mx); sum += z[e]; }
  float inv = 1.f / sum;
#pragma unroll
  for (int e = 0; e < E_; ++e) { z[e] *= inv; out_gw[(size_t)row * E_ + e] = z[e]; }
  int i0 = 0; float w0 = z[0];
#pragma unroll
  for (int e = 1; e < E_; ++e) if (z[e] > w0) { w0 = z[e]; i0 = e; }
  int i1 = -1; float w1 = -1e30f;
#pragma unroll
  for (int e = 0; e < E_; ++e) if (e != i0 && z[e] > w1) { w1 = z[e]; i1 = e; }
  float s2 = w0 + w1, inv2 = 1.f / s2;
  float t0 = w0 * inv2, t1 = w1 * inv2;
  out_tki[(size_t)row * 2 + 0] = (float)i0;
  out_tki[(size_t)row * 2 + 1] = (float)i1;
  out_tkw[(size_t)row * 2 + 0] = t0;
  out_tkw[(size_t)row * 2 + 1] = t1;
#pragma unroll
  for (int e = 0; e < E_; ++e)
    wfull[(size_t)row * E_ + e] = (e == i0) ? t0 : ((e == i1) ? t1 : 0.f);
}

// emb accumulation over experts: EMB += wfull[:,e] * EO  (e==0 initializes)
__global__ void combine_kernel(const float* __restrict__ eo, const float* __restrict__ wfull,
                               float* __restrict__ emb, int e) {
  long total = (long)B_ * D_;
  long i = (long)blockIdx.x * blockDim.x + threadIdx.x;
  long stride = (long)gridDim.x * blockDim.x;
  for (; i < total; i += stride) {
    long b = i / D_;
    float w = wfull[b * E_ + e];
    float v = w * eo[i];
    emb[i] = (e == 0) ? v : (emb[i] + v);
  }
}

// Classifier softmax (N=5) + argmax
__global__ __launch_bounds__(256)
void softmax5_kernel(const float* __restrict__ logits, float* __restrict__ probs,
                     float* __restrict__ pred_f, int* __restrict__ pred_i) {
  const int row = blockIdx.x * blockDim.x + threadIdx.x;
  if (row >= B_) return;
  float z[5], mx = -1e30f;
#pragma unroll
  for (int n = 0; n < 5; ++n) { z[n] = logits[(size_t)row * 5 + n]; mx = fmaxf(mx, z[n]); }
  float sum = 0.f;
#pragma unroll
  for (int n = 0; n < 5; ++n) { z[n] = __expf(z[n] - mx); sum += z[n]; }
  float inv = 1.f / sum;
  int am = 0; float best = -1.f;
#pragma unroll
  for (int n = 0; n < 5; ++n) {
    float p = z[n] * inv;
    probs[(size_t)row * 5 + n] = p;
    if (p > best) { best = p; am = n; }
  }
  pred_f[row] = (float)am;
  pred_i[row] = am;
}

// sf[s,b,:] *= (pred[b]==s)
__global__ void mask_sf_kernel(float* __restrict__ sf, const int* __restrict__ pred) {
  long total = (long)5 * B_ * D_;
  long i = (long)blockIdx.x * blockDim.x + threadIdx.x;
  long stride = (long)gridDim.x * blockDim.x;
  for (; i < total; i += stride) {
    int s = (int)(i / ((long)B_ * D_));
    int b = (int)((i / D_) % B_);
    sf[i] *= (pred[b] == s) ? 1.f : 0.f;
  }
}

// expert_usage = mean_b gate_weights (single block)
__global__ __launch_bounds__(256)
void usage_kernel(const float* __restrict__ gw, float* __restrict__ usage) {
  const int tid = threadIdx.x;
  float loc[E_];
#pragma unroll
  for (int e = 0; e < E_; ++e) loc[e] = 0.f;
  for (int b = tid; b < B_; b += 256)
#pragma unroll
    for (int e = 0; e < E_; ++e) loc[e] += gw[(size_t)b * E_ + e];
  __shared__ float red[256];
  for (int e = 0; e < E_; ++e) {
    red[tid] = loc[e];
    __syncthreads();
    for (int off = 128; off > 0; off >>= 1) {
      if (tid < off) red[tid] += red[tid + off];
      __syncthreads();
    }
    if (tid == 0) usage[e] = red[0] / (float)B_;
    __syncthreads();
  }
}

// ---------------------------------------------------------------------------
extern "C" void kernel_launch(void* const* d_in, const int* in_sizes, int n_in,
                              void* d_out, int out_size, void* d_ws, size_t ws_size,
                              hipStream_t stream) {
  (void)in_sizes; (void)n_in; (void)out_size; (void)ws_size;
  int ii = 0;
  const float* vis = (const float*)d_in[ii++];
  const float* mem = (const float*)d_in[ii++];
  const float* vp_w1 = (const float*)d_in[ii++]; const float* vp_b1 = (const float*)d_in[ii++];
  const float* vp_g1 = (const float*)d_in[ii++]; const float* vp_be1 = (const float*)d_in[ii++];
  const float* vp_w2 = (const float*)d_in[ii++]; const float* vp_b2 = (const float*)d_in[ii++];
  const float* vp_g2 = (const float*)d_in[ii++]; const float* vp_be2 = (const float*)d_in[ii++];
  const float* mp_w = (const float*)d_in[ii++]; const float* mp_b = (const float*)d_in[ii++];
  const float* mp_g = (const float*)d_in[ii++]; const float* mp_be = (const float*)d_in[ii++];
  const float* fc_w = (const float*)d_in[ii++]; const float* fc_b = (const float*)d_in[ii++];
  const float* fc_g = (const float*)d_in[ii++]; const float* fc_be = (const float*)d_in[ii++];
  const float* gate_w1 = (const float*)d_in[ii++]; const float* gate_b1 = (const float*)d_in[ii++];
  const float* gate_g = (const float*)d_in[ii++]; const float* gate_be = (const float*)d_in[ii++];
  const float* gate_w2 = (const float*)d_in[ii++]; const float* gate_b2 = (const float*)d_in[ii++];
  const float* ex_w1 = (const float*)d_in[ii++]; const float* ex_b1 = (const float*)d_in[ii++];
  const float* ex_g1 = (const float*)d_in[ii++]; const float* ex_be1 = (const float*)d_in[ii++];
  const float* ex_w2 = (const float*)d_in[ii++]; const float* ex_b2 = (const float*)d_in[ii++];
  const float* ex_g2 = (const float*)d_in[ii++]; const float* ex_be2 = (const float*)d_in[ii++];
  const float* ex_w3 = (const float*)d_in[ii++]; const float* ex_b3 = (const float*)d_in[ii++];
  const float* op_w = (const float*)d_in[ii++]; const float* op_b = (const float*)d_in[ii++];
  const float* cl_w1 = (const float*)d_in[ii++]; const float* cl_b1 = (const float*)d_in[ii++];
  const float* cl_g = (const float*)d_in[ii++]; const float* cl_be = (const float*)d_in[ii++];
  const float* cl_w2 = (const float*)d_in[ii++]; const float* cl_b2 = (const float*)d_in[ii++];
  const float* st_w = (const float*)d_in[ii++]; const float* st_b = (const float*)d_in[ii++];

  // d_out layout (return-order flat)
  float* o = (float*)d_out;
  float* out_emb    = o;                         // B*D
  float* out_logits = out_emb + (size_t)B_ * D_; // B*5
  float* out_probs  = out_logits + (size_t)B_ * 5;
  float* out_pred   = out_probs + (size_t)B_ * 5;     // B
  float* out_sf     = out_pred + B_;                  // 5*B*D
  float* out_gw     = out_sf + (size_t)5 * B_ * D_;   // B*E
  float* out_tki    = out_gw + (size_t)B_ * E_;       // B*2
  float* out_tkw    = out_tki + (size_t)B_ * 2;       // B*2
  float* out_usage  = out_tkw + (size_t)B_ * 2;       // E

  // workspace carve-out
  char* ws = (char*)d_ws;
  auto take = [&](size_t bytes) -> void* {
    void* p = (void*)ws;
    ws += (bytes + 255) & ~(size_t)255;
    return p;
  };
  typedef unsigned short u16;
  u16* Wvp1t = (u16*)take((size_t)H_ * V_ * 2);
  u16* Wvp2t = (u16*)take((size_t)H_ * H_ * 2);
  u16* Wmpt  = (u16*)take((size_t)H_ * MM_ * 2);
  u16* Wfct  = (u16*)take((size_t)H_ * 2 * H_ * 2);
  u16* Wg1t  = (u16*)take((size_t)H_ * H_ * 2);
  u16* Wex1t = (u16*)take((size_t)E_ * H_ * H_ * 2);
  u16* Wex2t = (u16*)take((size_t)E_ * H_ * H_ * 2);
  u16* Wex3t = (u16*)take((size_t)E_ * D_ * H_ * 2);
  u16* Wopt  = (u16*)take((size_t)D_ * D_ * 2);
  u16* Wcl1t = (u16*)take((size_t)128 * D_ * 2);
  u16* Wstt  = (u16*)take((size_t)5 * D_ * D_ * 2);
  u16* Xvis  = (u16*)take((size_t)B_ * V_ * 2);
  u16* Xmem  = (u16*)take((size_t)B_ * MM_ * 2);
  u16* bufA  = (u16*)take((size_t)B_ * 2 * H_ * 2);  // concat [v|m]
  u16* bufB  = (u16*)take((size_t)B_ * H_ * 2);      // combined c
  u16* bufC  = (u16*)take((size_t)B_ * H_ * 2);      // gate hidden / expert hidden
  u16* bufD  = (u16*)take((size_t)B_ * D_ * 2);      // emb bf16
  u16* bufE  = (u16*)take((size_t)B_ * 128 * 2);     // classifier hidden
  float* G32   = (float*)take((size_t)B_ * H_ * 4);
  float* EO32  = (float*)take((size_t)B_ * D_ * 4);
  float* EMB32 = (float*)take((size_t)B_ * D_ * 4);
  float* GATE32 = (float*)take((size_t)B_ * E_ * 4);
  float* WFULL  = (float*)take((size_t)B_ * E_ * 4);
  int*   PREDI  = (int*)take((size_t)B_ * 4);

  auto cvt = [&](const float* s, u16* d, long n) {
    cvt_kernel<<<8192, 256, 0, stream>>>(s, d, n);
  };
  auto cvtT = [&](const float* s, u16* d, int K, int N) {
    long total = (long)K * N;
    int blocks = (int)((total + 255) / 256);
    if (blocks > 16384) blocks = 16384;
    cvtT_kernel<<<blocks, 256, 0, stream>>>(s, d, K, N);
  };
  auto gemm = [&](const u16* A, int lda, const u16* Bt, int ldb, const float* bias,
                  float* C, int N, int K) {
    dim3 g(N / BN, B_ / BM);
    gemm_bf16_wmma<<<g, 256, 0, stream>>>(A, lda, Bt, ldb, bias, C, N, K);
  };
  auto lngelu = [&](const float* X, int N, const float* g, const float* b,
                    u16* Y, int ldy) {
    ln_gelu_kernel<<<B_, 256, 0, stream>>>(X, N, g, b, Y, ldy);
  };

  // --- weight conversion (f32 -> bf16, transposed to [N][K]) ---
  cvtT(vp_w1, Wvp1t, V_, H_);
  cvtT(vp_w2, Wvp2t, H_, H_);
  cvtT(mp_w, Wmpt, MM_, H_);
  cvtT(fc_w, Wfct, 2 * H_, H_);
  cvtT(gate_w1, Wg1t, H_, H_);
  for (int e = 0; e < E_; ++e) {
    cvtT(ex_w1 + (size_t)e * H_ * H_, Wex1t + (size_t)e * H_ * H_, H_, H_);
    cvtT(ex_w2 + (size_t)e * H_ * H_, Wex2t + (size_t)e * H_ * H_, H_, H_);
    cvtT(ex_w3 + (size_t)e * H_ * D_, Wex3t + (size_t)e * D_ * H_, H_, D_);
  }
  cvtT(op_w, Wopt, D_, D_);
  cvtT(cl_w1, Wcl1t, D_, 128);
  for (int s = 0; s < 5; ++s)
    cvtT(st_w + (size_t)s * D_ * D_, Wstt + (size_t)s * D_ * D_, D_, D_);
  cvt(vis, Xvis, (long)B_ * V_);
  cvt(mem, Xmem, (long)B_ * MM_);

  // --- vision processor ---
  gemm(Xvis, V_, Wvp1t, V_, vp_b1, G32, H_, V_);
  lngelu(G32, H_, vp_g1, vp_be1, bufB, H_);
  gemm(bufB, H_, Wvp2t, H_, vp_b2, G32, H_, H_);
  lngelu(G32, H_, vp_g2, vp_be2, bufA, 2 * H_);            // left half of concat
  // --- memory processor ---
  gemm(Xmem, MM_, Wmpt, MM_, mp_b, G32, H_, MM_);
  lngelu(G32, H_, mp_g, mp_be, bufA + H_, 2 * H_);         // right half of concat
  // --- feature combiner ---
  gemm(bufA, 2 * H_, Wfct, 2 * H_, fc_b, G32, H_, 2 * H_);
  lngelu(G32, H_, fc_g, fc_be, bufB, H_);                  // c (kept for experts)
  // --- gate ---
  gemm(bufB, H_, Wg1t, H_, gate_b1, G32, H_, H_);
  lngelu(G32, H_, gate_g, gate_be, bufC, H_);
  small_gemm<E_><<<B_ / 256, 256, 0, stream>>>(bufC, H_, gate_w2, gate_b2, GATE32);
  gate_kernel<<<B_ / 256, 256, 0, stream>>>(GATE32, out_gw, out_tki, out_tkw, WFULL);
  usage_kernel<<<1, 256, 0, stream>>>(out_gw, out_usage);

  // --- experts (sequential, dense, accumulated by routing weights) ---
  for (int e = 0; e < E_; ++e) {
    gemm(bufB, H_, Wex1t + (size_t)e * H_ * H_, H_, ex_b1 + (size_t)e * H_, G32, H_, H_);
    lngelu(G32, H_, ex_g1 + (size_t)e * H_, ex_be1 + (size_t)e * H_, bufC, H_);
    gemm(bufC, H_, Wex2t + (size_t)e * H_ * H_, H_, ex_b2 + (size_t)e * H_, G32, H_, H_);
    lngelu(G32, H_, ex_g2 + (size_t)e * H_, ex_be2 + (size_t)e * H_, bufC, H_);
    gemm(bufC, H_, Wex3t + (size_t)e * D_ * H_, H_, ex_b3 + (size_t)e * D_, EO32, D_, H_);
    combine_kernel<<<8192, 256, 0, stream>>>(EO32, WFULL, EMB32, e);
  }

  // --- output projection -> emb ---
  cvt(EMB32, bufD, (long)B_ * D_);
  gemm(bufD, D_, Wopt, D_, op_b, out_emb, D_, D_);
  cvt(out_emb, bufD, (long)B_ * D_);                       // emb in bf16

  // --- classifier ---
  gemm(bufD, D_, Wcl1t, D_, cl_b1, EO32, 128, D_);
  lngelu(EO32, 128, cl_g, cl_be, bufE, 128);
  small_gemm<5><<<B_ / 256, 256, 0, stream>>>(bufE, 128, cl_w2, cl_b2, out_logits);
  softmax5_kernel<<<B_ / 256, 256, 0, stream>>>(out_logits, out_probs, out_pred, PREDI);

  // --- state extractors + mask ---
  for (int s = 0; s < 5; ++s)
    gemm(bufD, D_, Wstt + (size_t)s * D_ * D_, D_, st_b + (size_t)s * D_,
         out_sf + (size_t)s * B_ * D_, D_, D_);
  mask_sf_kernel<<<8192, 256, 0, stream>>>(out_sf, PREDI);
}
